// Model_15668040696444
// MI455X (gfx1250) — matrix operand order repaired and checked by reading, not compiled, not run on hardware
//
#include <hip/hip_runtime.h>

// ---------------------------------------------------------------------------
// Graph-transformer pointer network forward for MI455X (gfx1250, wave32).
// Heavy GEMMs use v_wmma_f32_16x16x32_f16; decode loop uses streaming
// LDS-reduction kernels (memory-bound part: ~25 GB over 127 steps).
// ---------------------------------------------------------------------------

typedef __attribute__((ext_vector_type(16))) _Float16 v16h;
typedef __attribute__((ext_vector_type(8)))  _Float16 v8h;
typedef __attribute__((ext_vector_type(8)))  float    v8f;

#define DMODEL 256
#define NNODE  128
#define SQD    16
#define NHEAD  8
#define FFD    512
#define NDSET  4
#define BSZ    64
#define TSTEP  127
#define NPAIR  16384   // NNODE*NNODE
#define LNEPS  1e-5f

// ===========================================================================
// Generic WMMA GEMM:  C[z] = act(alpha * A[z] @ Bt[z]^T + bias)
//   A : f16, M x K, row stride lda
//   Bt: f16, N x K, row stride ldb  (i.e. computes A @ B with B = Bt^T)
//   C : f32 (ldc), optional Ch f16 mirror (ldch)
//   per-z offset = (z/nInner)*out + (z%nInner)*in   (element offsets)
// Requires M%16==0, N%16==0, K%32==0, 16B-aligned rows (lda,ldb mult of 8/16).
// One wave (32 threads) per 16x16 output tile. grid = (M/16, N/16, Z).
// ===========================================================================
__global__ void gemm_bt_wmma(const _Float16* __restrict__ A,
                             const _Float16* __restrict__ Bt,
                             const float* __restrict__ bias,
                             float* __restrict__ C,
                             _Float16* __restrict__ Ch,
                             int K, int lda, int ldb, int ldc, int ldch,
                             int nInner,
                             long aOut, long aIn, long bOut, long bIn,
                             long cOut, long cIn,
                             float alpha, int relu)
{
    const int z   = blockIdx.z;
    const long zo = z / nInner;
    const long zi = z % nInner;
    const int lane = threadIdx.x;      // 0..31
    const int r    = lane & 15;
    const int hi   = lane >> 4;

    const _Float16* Ap = A  + zo*aOut + zi*aIn + (long)(blockIdx.x*16 + r) * lda;
    const _Float16* Bp = Bt + zo*bOut + zi*bIn + (long)(blockIdx.y*16 + r) * ldb;

    v8f acc = {};
    for (int k0 = 0; k0 < K; k0 += 32) {
        // A operand, contiguous k order chosen here (B matches): element h -> K = k0 + hi*16 + h
        v8h alo = *(const v8h*)(Ap + k0 + hi*16);
        v8h ahi = *(const v8h*)(Ap + k0 + hi*16 + 8);
        // B operand, same contiguous k order as A: element h -> K = k0 + hi*16 + h
        v16h b = *(const v16h*)(Bp + k0 + hi*16);
        v16h a;
#pragma unroll
        for (int i = 0; i < 8; ++i) { a[i] = alo[i]; a[i+8] = ahi[i]; }
        acc = __builtin_amdgcn_wmma_f32_16x16x32_f16(false, a, false, b,
                                                     (short)0, acc, false, false);
    }

    const int cn = blockIdx.y*16 + r;
    float*     Cp  = C + zo*cOut + zi*cIn;
    _Float16*  Chp = Ch ? (Ch + zo*cOut + zi*cIn) : (_Float16*)0;
    const float bv = bias ? bias[cn] : 0.f;
#pragma unroll
    for (int i = 0; i < 8; ++i) {
        const int cm = blockIdx.x*16 + i + 8*hi;   // C f32 layout: M = i + 8*hi
        float v = acc[i]*alpha + bv;
        if (relu) v = fmaxf(v, 0.f);
        Cp[(long)cm*ldc + cn] = v;
        if (Chp) Chp[(long)cm*ldch + cn] = (_Float16)v;
    }
}

// ---------------------------------------------------------------------------
// Pack weight W (fi x fo, f32, row-major) -> Wt (fo x ldb, f16, zero padded).
// grid = fo blocks, 128 threads.
// ---------------------------------------------------------------------------
__global__ void pack_wt(const float* __restrict__ W, _Float16* __restrict__ Wt,
                        int fi, int fo, int ldb)
{
    const int o = blockIdx.x;
    for (int i = threadIdx.x; i < ldb; i += blockDim.x)
        Wt[(long)o*ldb + i] = (_Float16)((i < fi) ? W[(long)i*fo + o] : 0.f);
}

// ---------------------------------------------------------------------------
// Embedding: (4,128,2) @ W(2,256) + b.  grid = 512 rows, 256 threads.
// ---------------------------------------------------------------------------
__global__ void embed_k(const float* __restrict__ x,
                        const float* __restrict__ wterm, const float* __restrict__ bterm,
                        const float* __restrict__ wnon,  const float* __restrict__ bnon,
                        const int* __restrict__ terminal,
                        float* __restrict__ Y, _Float16* __restrict__ Yh)
{
    const int row = blockIdx.x;            // d*128 + n
    const int n   = row & (NNODE-1);
    const int c   = threadIdx.x;
    const int term = *terminal;
    const float* W = (n < term) ? wterm : wnon;
    const float* B = (n < term) ? bterm : bnon;
    const float x0 = x[row*2], x1 = x[row*2+1];
    const float y = x0*W[c] + x1*W[DMODEL + c] + B[c];
    Y [(long)row*DMODEL + c] = y;
    Yh[(long)row*DMODEL + c] = (_Float16)y;
}

// ---------------------------------------------------------------------------
// Row softmax (128 cols) f32 -> f16.  grid = rows, 128 threads.
// ---------------------------------------------------------------------------
__global__ void softmax128(const float* __restrict__ S, _Float16* __restrict__ P)
{
    __shared__ float red[128];
    const long off = (long)blockIdx.x * 128;
    const int tid = threadIdx.x;
    const float v = S[off + tid];
    red[tid] = v; __syncthreads();
    for (int st = 64; st > 0; st >>= 1) { if (tid < st) red[tid] = fmaxf(red[tid], red[tid+st]); __syncthreads(); }
    const float m = red[0]; __syncthreads();
    const float e = __expf(v - m);
    red[tid] = e; __syncthreads();
    for (int st = 64; st > 0; st >>= 1) { if (tid < st) red[tid] += red[tid+st]; __syncthreads(); }
    P[off + tid] = (_Float16)(e / red[0]);
}

// ---------------------------------------------------------------------------
// Pack V^T per (d,h): Vt[z][c][r] = qkv[d*128+r][512 + h*32 + c], f16.
// grid = 32, 128 threads.
// ---------------------------------------------------------------------------
__global__ void pack_vt(const float* __restrict__ qkv, _Float16* __restrict__ Vt)
{
    const int z = blockIdx.x, d = z >> 3, h = z & 7;
    const int r = threadIdx.x;
    const float* src = qkv + ((long)(d*NNODE + r))*768 + 512 + h*32;
    _Float16*   dst = Vt + (long)z*32*NNODE;
    for (int c = 0; c < 32; ++c) dst[c*NNODE + r] = (_Float16)src[c];
}

// ---------------------------------------------------------------------------
// LayerNorm(residual): y = LN(X + R) * g + b.  grid = rows, 256 threads.
// ---------------------------------------------------------------------------
__global__ void lnorm_res(const float* __restrict__ X, const float* __restrict__ R,
                          const float* __restrict__ g, const float* __restrict__ b,
                          float* __restrict__ Y, _Float16* __restrict__ Yh)
{
    __shared__ float red[256];
    const long off = (long)blockIdx.x * DMODEL;
    const int tid = threadIdx.x;
    const float v = X[off + tid] + R[off + tid];
    red[tid] = v; __syncthreads();
    for (int st = 128; st > 0; st >>= 1) { if (tid < st) red[tid] += red[tid+st]; __syncthreads(); }
    const float mean = red[0] * (1.f/DMODEL); __syncthreads();
    const float dv = v - mean;
    red[tid] = dv*dv; __syncthreads();
    for (int st = 128; st > 0; st >>= 1) { if (tid < st) red[tid] += red[tid+st]; __syncthreads(); }
    const float var = red[0] * (1.f/DMODEL);
    const float y = dv * rsqrtf(var + LNEPS) * g[tid] + b[tid];
    Y [off + tid] = y;
    Yh[off + tid] = (_Float16)y;
}

// ---------------------------------------------------------------------------
// BatchNorm over (0,1) for (512,256).  grid = 256 channels, 256 threads.
// ---------------------------------------------------------------------------
__global__ void bnorm_enc(const float* __restrict__ X,
                          const float* __restrict__ g, const float* __restrict__ b,
                          float* __restrict__ Y)
{
    __shared__ float red[256];
    const int c = blockIdx.x, tid = threadIdx.x;
    const float v0 = X[(long)tid*DMODEL + c];
    const float v1 = X[(long)(tid+256)*DMODEL + c];
    red[tid] = v0 + v1; __syncthreads();
    for (int st = 128; st > 0; st >>= 1) { if (tid < st) red[tid] += red[tid+st]; __syncthreads(); }
    const float mean = red[0] * (1.f/512.f); __syncthreads();
    const float d0 = v0-mean, d1 = v1-mean;
    red[tid] = d0*d0 + d1*d1; __syncthreads();
    for (int st = 128; st > 0; st >>= 1) { if (tid < st) red[tid] += red[tid+st]; __syncthreads(); }
    const float inv = rsqrtf(red[0] * (1.f/512.f) + LNEPS);
    Y[(long)tid*DMODEL + c]       = d0*inv*g[c] + b[c];
    Y[(long)(tid+256)*DMODEL + c] = d1*inv*g[c] + b[c];
}

// ---------------------------------------------------------------------------
// Edge feature: [v_i - v_j, max(v_i,v_j)] -> f16 (65536 x 512).
// grid = 65536 pairs, 256 threads.
// ---------------------------------------------------------------------------
__global__ void edge_feat(const float* __restrict__ venc, _Float16* __restrict__ E)
{
    const long pair = blockIdx.x;                // d*16384 + i*128 + j
    const int d = (int)(pair >> 14);
    const int rem = (int)(pair & 16383);
    const int i = rem >> 7, j = rem & 127;
    const int c = threadIdx.x;
    const float vi = venc[((long)d*NNODE + i)*DMODEL + c];
    const float vj = venc[((long)d*NNODE + j)*DMODEL + c];
    _Float16* row = E + pair*512;
    row[c]          = (_Float16)(vi - vj);
    row[DMODEL + c] = (_Float16)fmaxf(vi, vj);
}

// ---------------------------------------------------------------------------
// Mean over nodes -> padded f16 A (16 x 256). grid = 4, 256 threads.
// ---------------------------------------------------------------------------
__global__ void mean_rows(const float* __restrict__ venc, _Float16* __restrict__ A)
{
    const int d = blockIdx.x, c = threadIdx.x;
    float s = 0.f;
    for (int n = 0; n < NNODE; ++n) s += venc[((long)d*NNODE + n)*DMODEL + c];
    A[(long)d*DMODEL + c] = (_Float16)(s * (1.f/NNODE));
}

// ---------------------------------------------------------------------------
// Small 16x16 GEMM: Y(64,16) = act(X(64,16) @ W(16,16) + b). grid=64, 16 thr.
// ---------------------------------------------------------------------------
__global__ void gemm16(const float* __restrict__ X, int ldx,
                       const float* __restrict__ W, const float* __restrict__ B,
                       float* __restrict__ Y, int ldy, int relu)
{
    const int bb = blockIdx.x, n = threadIdx.x;
    const float* x = X + (long)bb*ldx;
    float acc = B[n];
    for (int k = 0; k < SQD; ++k) acc += x[k]*W[k*SQD + n];
    if (relu) acc = fmaxf(acc, 0.f);
    Y[(long)bb*ldy + n] = acc;
}

// ---------------------------------------------------------------------------
// h = maxpool2( concat(v_mean[b], s[b]) ). grid=64, 16 threads.
// ---------------------------------------------------------------------------
__global__ void dec_h_k(const float* __restrict__ vmean, const float* __restrict__ s,
                        float* __restrict__ h)
{
    const int bb = blockIdx.x, i = threadIdx.x, d = bb >> 4;
    float a, b;
    if (i < 8) { a = vmean[d*SQD + 2*i];       b = vmean[d*SQD + 2*i + 1]; }
    else       { a = s[bb*SQD + 2*(i-8)];      b = s[bb*SQD + 2*(i-8) + 1]; }
    h[bb*SQD + i] = fmaxf(a, b);
}

// ---------------------------------------------------------------------------
// Decoder self-attention (hd=2, causal up to t). grid=64, 8 threads (heads).
// ---------------------------------------------------------------------------
__global__ void mha_self(const float* __restrict__ q, const float* __restrict__ Ksa,
                         const float* __restrict__ Vsa, int t, float* __restrict__ o)
{
    const int bb = blockIdx.x, h = threadIdx.x;
    const float q0 = q[bb*SQD + h*2], q1 = q[bb*SQD + h*2 + 1];
    const float inv = 0.70710678f;
    float m = -1e30f;
    for (int j = 0; j <= t; ++j) {
        const float* kv = Ksa + ((long)bb*TSTEP + j)*SQD + h*2;
        m = fmaxf(m, (q0*kv[0] + q1*kv[1])*inv);
    }
    float sum = 0.f, a0 = 0.f, a1 = 0.f;
    for (int j = 0; j <= t; ++j) {
        const float* kv = Ksa + ((long)bb*TSTEP + j)*SQD + h*2;
        const float* vv = Vsa + ((long)bb*TSTEP + j)*SQD + h*2;
        const float e = __expf((q0*kv[0] + q1*kv[1])*inv - m);
        sum += e; a0 += e*vv[0]; a1 += e*vv[1];
    }
    o[bb*SQD + h*2]     = a0/sum;
    o[bb*SQD + h*2 + 1] = a1/sum;
}

// ---------------------------------------------------------------------------
// Decoder cross-attention over 16384 positions (hd=2, masked).
// grid = (8 heads, 64 batch), 256 threads, LDS reductions. Memory-bound.
// ---------------------------------------------------------------------------
__global__ void mha_cross(const float* __restrict__ q, const float* __restrict__ katt,
                          const float* __restrict__ vatt, const int* __restrict__ masks,
                          float* __restrict__ o)
{
    __shared__ float red[256];
    const int h = blockIdx.x, bb = blockIdx.y, tid = threadIdx.x;
    const int d = bb >> 4;
    const float q0 = q[bb*SQD + h*2], q1 = q[bb*SQD + h*2 + 1];
    const float inv = 0.70710678f;
    const float* Kd = katt + (long)d*NPAIR*48;
    const float* Vd = vatt + (long)d*NPAIR*32;
    const int*   mk = masks + (long)bb*NPAIR;

    float m = -1e30f;
    for (int p = tid; p < NPAIR; p += 256) {
        __builtin_prefetch(Kd + (long)(p + 1024)*48, 0, 1);
        const float s = (mk[p] > 0) ? -1e9f
                      : (q0*Kd[(long)p*48 + h*2] + q1*Kd[(long)p*48 + h*2 + 1])*inv;
        m = fmaxf(m, s);
    }
    red[tid] = m; __syncthreads();
    for (int st = 128; st > 0; st >>= 1) { if (tid < st) red[tid] = fmaxf(red[tid], red[tid+st]); __syncthreads(); }
    m = red[0]; __syncthreads();

    float sum = 0.f, a0 = 0.f, a1 = 0.f;
    for (int p = tid; p < NPAIR; p += 256) {
        const float s = (mk[p] > 0) ? -1e9f
                      : (q0*Kd[(long)p*48 + h*2] + q1*Kd[(long)p*48 + h*2 + 1])*inv;
        const float e = __expf(s - m);
        sum += e;
        a0 += e*Vd[(long)p*32 + h*2];
        a1 += e*Vd[(long)p*32 + h*2 + 1];
    }
    red[tid] = sum; __syncthreads();
    for (int st = 128; st > 0; st >>= 1) { if (tid < st) red[tid] += red[tid+st]; __syncthreads(); }
    const float S = red[0]; __syncthreads();
    red[tid] = a0; __syncthreads();
    for (int st = 128; st > 0; st >>= 1) { if (tid < st) red[tid] += red[tid+st]; __syncthreads(); }
    const float A0 = red[0]; __syncthreads();
    red[tid] = a1; __syncthreads();
    for (int st = 128; st > 0; st >>= 1) { if (tid < st) red[tid] += red[tid+st]; __syncthreads(); }
    if (tid == 0) { o[bb*SQD + h*2] = A0/S; o[bb*SQD + h*2 + 1] = red[0]/S; }
}

// ---------------------------------------------------------------------------
// BatchNorm over batch for (64,16) with residual, in-place capable.
// grid = 16 channels, 64 threads.
// ---------------------------------------------------------------------------
__global__ void bnorm_res16(float* __restrict__ H, const float* __restrict__ O,
                            const float* __restrict__ g, const float* __restrict__ b)
{
    __shared__ float red[64];
    const int c = blockIdx.x, tid = threadIdx.x;
    const float v = H[tid*SQD + c] + O[tid*SQD + c];
    red[tid] = v; __syncthreads();
    for (int st = 32; st > 0; st >>= 1) { if (tid < st) red[tid] += red[tid+st]; __syncthreads(); }
    const float mean = red[0] * (1.f/BSZ); __syncthreads();
    const float dv = v - mean;
    red[tid] = dv*dv; __syncthreads();
    for (int st = 32; st > 0; st >>= 1) { if (tid < st) red[tid] += red[tid+st]; __syncthreads(); }
    H[tid*SQD + c] = dv * rsqrtf(red[0]*(1.f/BSZ) + LNEPS) * g[c] + b[c];
}

// ---------------------------------------------------------------------------
// Final pointer scores: 10*tanh(qf . Katt[:,16:32] / 4), masked; fused
// argmax + log-softmax-at-argmax. grid=64, 256 threads.
// ---------------------------------------------------------------------------
__global__ void final_step(const float* __restrict__ qf, const float* __restrict__ katt,
                           const int* __restrict__ masks,
                           int* __restrict__ idx_out, float* __restrict__ logp_out)
{
    __shared__ float rv[256];
    __shared__ int   ri[256];
    const int bb = blockIdx.x, tid = threadIdx.x, d = bb >> 4;
    const float* Kd = katt + (long)d*NPAIR*48 + 16;
    const int*   mk = masks + (long)bb*NPAIR;
    float qv[SQD];
    for (int c = 0; c < SQD; ++c) qv[c] = qf[bb*SQD + c];

    float m = -1e30f; int mi = 0;
    for (int p = tid; p < NPAIR; p += 256) {
        float s;
        if (mk[p] > 0) s = -1e9f;
        else {
            const float* kp = Kd + (long)p*48;
            float acc = 0.f;
            for (int c = 0; c < SQD; ++c) acc += qv[c]*kp[c];
            s = 10.f * tanhf(acc * 0.25f);
        }
        if (s > m) { m = s; mi = p; }
    }
    rv[tid] = m; ri[tid] = mi; __syncthreads();
    for (int st = 128; st > 0; st >>= 1) {
        if (tid < st && rv[tid+st] > rv[tid]) { rv[tid] = rv[tid+st]; ri[tid] = ri[tid+st]; }
        __syncthreads();
    }
    m = rv[0]; const int am = ri[0]; __syncthreads();

    float sum = 0.f;
    for (int p = tid; p < NPAIR; p += 256) {
        float s;
        if (mk[p] > 0) s = -1e9f;
        else {
            const float* kp = Kd + (long)p*48;
            float acc = 0.f;
            for (int c = 0; c < SQD; ++c) acc += qv[c]*kp[c];
            s = 10.f * tanhf(acc * 0.25f);
        }
        sum += __expf(s - m);
    }
    rv[tid] = sum; __syncthreads();
    for (int st = 128; st > 0; st >>= 1) { if (tid < st) rv[tid] += rv[tid+st]; __syncthreads(); }
    if (tid == 0) { idx_out[bb] = am; logp_out[bb] = -__logf(rv[0]); }
}

// ---------------------------------------------------------------------------
// Decode-state init. grid=64, 128 threads.
// ---------------------------------------------------------------------------
__global__ void dec_init(const int* __restrict__ init_nodes, const float* __restrict__ E,
                         int* masks, int* degrees, float* s,
                         int* alive, int* lengths, float* lsum)
{
    const int bb = blockIdx.x, i = threadIdx.x;
    const int node = init_nodes[bb];
    for (int j = 0; j < NNODE; ++j) {
        int v = 1 + (i == j ? 1 : 0);
        if (i == node) v -= 1;       // masks[b, node, :] -= 1
        if (j == node) v += 1;       // masks[b, :, node] += 1
        masks[((long)bb*NNODE + i)*NNODE + j] = v;
    }
    degrees[bb*NNODE + i] = 0;
    if (i < SQD) {
        const int d = bb >> 4;
        s[bb*SQD + i] = E[((long)d*NPAIR + (long)node*NNODE + node)*SQD + i];
    }
    if (i == 0) { alive[bb] = 1; lengths[bb] = 1; lsum[bb] = 0.f; }
}

// ---------------------------------------------------------------------------
// Decode-state update after argmax. grid=64, 128 threads.
// ---------------------------------------------------------------------------
__global__ void dec_update(const int* __restrict__ idx, const float* __restrict__ logp,
                           const float* __restrict__ E,
                           int* masks, int* degrees, float* s,
                           int* alive, int* lengths, float* lsum,
                           float* __restrict__ out, int t, const int* __restrict__ terminal)
{
    __shared__ int allpos;
    const int bb = blockIdx.x, i = threadIdx.x;
    const int id = idx[bb];
    const int left = id / NNODE, right = id % NNODE;
    const int av = alive[bb];
    if (i == 0) {
        allpos = 1;
        out[bb*(2*TSTEP) + t]         = (float)(left * av);
        out[bb*(2*TSTEP) + TSTEP + t] = (float)(right * av);
        lsum[bb] += logp[bb] * (float)av;
        degrees[bb*NNODE + left]  += 1;
        degrees[bb*NNODE + right] += 1;
    }
    if (i < SQD) {
        const int d = bb >> 4;
        s[bb*SQD + i] = E[((long)d*NPAIR + (long)left*NNODE + right)*SQD + i];
    }
    masks[((long)bb*NNODE + right)*NNODE + i] -= 1;
    masks[((long)bb*NNODE + i)*NNODE + right] += 1;
    __syncthreads();
    const int term = *terminal;
    if (i < term && degrees[bb*NNODE + i] <= 0) allpos = 0;
    __syncthreads();
    if (i == 0) {
        const int na = (av != 0) && (allpos == 0);
        alive[bb] = na;
        lengths[bb] += na;
    }
}

__global__ void finalize_k(const float* __restrict__ lsum, const int* __restrict__ lengths,
                           float* __restrict__ out)
{
    const int i = threadIdx.x;
    out[BSZ*2*TSTEP + i]       = lsum[i];
    out[BSZ*2*TSTEP + BSZ + i] = (float)lengths[i];
}

// ===========================================================================
// Host side
// ===========================================================================
static inline void gemmL(hipStream_t st, const _Float16* A, const _Float16* Bt,
                         const float* bias, float* C, _Float16* Ch,
                         int M, int N, int K, int lda, int ldb, int ldc, int ldch,
                         int Z, int nInner, long aOut, long aIn, long bOut, long bIn,
                         long cOut, long cIn, float alpha, int relu)
{
    dim3 g(M/16, N/16, Z);
    gemm_bt_wmma<<<g, dim3(32), 0, st>>>(A, Bt, bias, C, Ch, K, lda, ldb, ldc, ldch,
                                         nInner, aOut, aIn, bOut, bIn, cOut, cIn,
                                         alpha, relu);
}

extern "C" void kernel_launch(void* const* d_in, const int* in_sizes, int n_in,
                              void* d_out, int out_size, void* d_ws, size_t ws_size,
                              hipStream_t stream)
{
    (void)in_sizes; (void)n_in; (void)out_size;
    // ---- param leaves: jax tree order (dict keys sorted at each level) ----
    int q = 1;  // d_in[0] = x
    const float* P[80];
    for (int i = 0; i < 80; ++i) P[i] = (const float*)0;
    #define NEXT ((const float*)d_in[q++])
    const float* x = (const float*)d_in[0];
    const float* p_bnb = NEXT; const float* p_bng = NEXT;
    // dec0 (sorted): bn_at_b, bn_at_g, bn_mlp_b, bn_mlp_g, bn_sa_b, bn_sa_g,
    //                w0_at{b,w}, w0_sa{b,w}, w1{b,w}, w2{b,w},
    //                wk_sa{b,w}, wq_at{b,w}, wq_sa{b,w}, wv_sa{b,w}
    const float* bn_at_b = NEXT; const float* bn_at_g = NEXT;
    const float* bn_mlp_b = NEXT; const float* bn_mlp_g = NEXT;
    const float* bn_sa_b = NEXT; const float* bn_sa_g = NEXT;
    const float* w0at_b = NEXT; const float* w0at_w = NEXT;
    const float* w0sa_b = NEXT; const float* w0sa_w = NEXT;
    const float* w1_b = NEXT; const float* w1_w = NEXT;
    const float* w2_b = NEXT; const float* w2_w = NEXT;
    const float* wk_b = NEXT; const float* wk_w = NEXT;
    const float* wqat_b = NEXT; const float* wqat_w = NEXT;
    const float* wqsa_b = NEXT; const float* wqsa_w = NEXT;
    const float* wv_b = NEXT; const float* wv_w = NEXT;
    // enc[l] (sorted): ff1{b,w}, ff2{b,w}, ln1b, ln1g, ln2b, ln2g, out{b,w}, qkv{b,w}
    const float *ff1b[3], *ff1w[3], *ff2b[3], *ff2w[3], *ln1b[3], *ln1g[3],
                *ln2b[3], *ln2g[3], *outb[3], *outw[3], *qkvb[3], *qkvw[3];
    for (int l = 0; l < 3; ++l) {
        ff1b[l] = NEXT; ff1w[l] = NEXT; ff2b[l] = NEXT; ff2w[l] = NEXT;
        ln1b[l] = NEXT; ln1g[l] = NEXT; ln2b[l] = NEXT; ln2g[l] = NEXT;
        outb[l] = NEXT; outw[l] = NEXT; qkvb[l] = NEXT; qkvw[l] = NEXT;
    }
    const float* line_b = NEXT; const float* line_w = NEXT;
    const float* ling_b = NEXT; const float* ling_w = NEXT;
    const float* link_b = NEXT; const float* link_w = NEXT;
    const float* linv_b = NEXT; const float* linv_w = NEXT;
    const float* nont_b = NEXT; const float* nont_w = NEXT;
    const float* term_b = NEXT; const float* term_w = NEXT;
    const float* wqf_b = NEXT; const float* wqf_w = NEXT;
    #undef NEXT
    const int* init_nodes = (const int*)d_in[q++];
    const int* terminal   = (const int*)d_in[q++];
    float* out = (float*)d_out;

    // ---- workspace carve-up ----
    size_t off = 0;
    char* base = (char*)d_ws;
    #define ALLOC(T, name, count) \
        T* name = (T*)(base + off); off = (off + (size_t)(count)*sizeof(T) + 255) & ~(size_t)255;
    ALLOC(float,    xA,   512*DMODEL)   ALLOC(_Float16, xAh,  512*DMODEL)
    ALLOC(float,    xB,   512*DMODEL)   ALLOC(_Float16, xBh,  512*DMODEL)
    ALLOC(float,    qkv,  512*768)      ALLOC(_Float16, qkvh, 512*768)
    ALLOC(float,    scores, 32*NNODE*NNODE)
    ALLOC(_Float16, probsh, 32*NNODE*NNODE)
    ALLOC(_Float16, vth,  32*32*NNODE)
    ALLOC(float,    attno, 512*DMODEL)  ALLOC(_Float16, attnoh, 512*DMODEL)
    ALLOC(float,    t1,   512*FFD)      ALLOC(_Float16, t1h,  512*FFD)
    ALLOC(float,    t0,   512*DMODEL)
    ALLOC(float,    venc, 512*DMODEL)
    ALLOC(_Float16, Eh,   (size_t)NDSET*NPAIR*512)
    ALLOC(float,    eF,   (size_t)NDSET*NPAIR*SQD)
    ALLOC(_Float16, ehPad,(size_t)NDSET*NPAIR*32)
    ALLOC(float,    katt, (size_t)NDSET*NPAIR*48)
    ALLOC(float,    vatt, (size_t)NDSET*NPAIR*32)
    ALLOC(_Float16, vmAh, 16*DMODEL)    ALLOC(float, vmC, 16*16)
    ALLOC(_Float16, qkvT0, 768*DMODEL)  ALLOC(_Float16, qkvT1, 768*DMODEL)
    ALLOC(_Float16, qkvT2, 768*DMODEL)
    ALLOC(_Float16, outT0, DMODEL*DMODEL) ALLOC(_Float16, outT1, DMODEL*DMODEL)
    ALLOC(_Float16, outT2, DMODEL*DMODEL)
    ALLOC(_Float16, ff1T0, FFD*DMODEL)  ALLOC(_Float16, ff1T1, FFD*DMODEL)
    ALLOC(_Float16, ff1T2, FFD*DMODEL)
    ALLOC(_Float16, ff2T0, DMODEL*FFD)  ALLOC(_Float16, ff2T1, DMODEL*FFD)
    ALLOC(_Float16, ff2T2, DMODEL*FFD)
    ALLOC(_Float16, lineT, 16*512)      ALLOC(_Float16, lingT, 16*DMODEL)
    ALLOC(_Float16, linkT, 48*32)       ALLOC(_Float16, linvT, 32*32)
    ALLOC(int,      masks, (size_t)BSZ*NPAIR)
    ALLOC(int,      degrees, BSZ*NNODE)
    ALLOC(float,    sbuf, BSZ*SQD)
    ALLOC(float,    Ksa,  (size_t)BSZ*TSTEP*SQD)
    ALLOC(float,    Vsa,  (size_t)BSZ*TSTEP*SQD)
    ALLOC(float,    hbuf, BSZ*SQD)      ALLOC(float, qb, BSZ*SQD)
    ALLOC(float,    ob,   BSZ*SQD)      ALLOC(float, tb, BSZ*SQD)
    ALLOC(float,    mb,   BSZ*SQD)
    ALLOC(int,      idxb, BSZ)          ALLOC(float, logpb, BSZ)
    ALLOC(int,      alive, BSZ)         ALLOC(int, lens, BSZ)
    ALLOC(float,    lsum, BSZ)
    #undef ALLOC
    if (off > ws_size) return;   // workspace too small: bail deterministically

    _Float16* qkvT[3] = {qkvT0, qkvT1, qkvT2};
    _Float16* outT[3] = {outT0, outT1, outT2};
    _Float16* ff1T[3] = {ff1T0, ff1T1, ff1T2};
    _Float16* ff2T[3] = {ff2T0, ff2T1, ff2T2};

    // zero-padded regions
    hipMemsetAsync(ehPad, 0, (size_t)NDSET*NPAIR*32*sizeof(_Float16), stream);
    hipMemsetAsync(vmAh, 0, 16*DMODEL*sizeof(_Float16), stream);

    // ---- pack weights to f16 (transposed, K-padded) ----
    for (int l = 0; l < 3; ++l) {
        pack_wt<<<768,  128, 0, stream>>>(qkvw[l], qkvT[l], DMODEL, 768, DMODEL);
        pack_wt<<<DMODEL,128, 0, stream>>>(outw[l], outT[l], DMODEL, DMODEL, DMODEL);
        pack_wt<<<FFD,  128, 0, stream>>>(ff1w[l], ff1T[l], DMODEL, FFD, DMODEL);
        pack_wt<<<DMODEL,128, 0, stream>>>(ff2w[l], ff2T[l], FFD, DMODEL, FFD);
    }
    pack_wt<<<16, 128, 0, stream>>>(line_w, lineT, 512, 16, 512);
    pack_wt<<<16, 128, 0, stream>>>(ling_w, lingT, DMODEL, 16, DMODEL);
    pack_wt<<<48, 128, 0, stream>>>(link_w, linkT, SQD, 48, 32);
    pack_wt<<<32, 128, 0, stream>>>(linv_w, linvT, SQD, 32, 32);

    // ---- embedding ----
    embed_k<<<512, DMODEL, 0, stream>>>(x, term_w, term_b, nont_w, nont_b,
                                        terminal, xA, xAh);

    // ---- encoder (3 layers) ----
    const float isq32 = 0.17677669529663687f;  // 1/sqrt(32)
    for (int l = 0; l < 3; ++l) {
        // QKV: (512,768) = xAh(512,256) @ Wqkv
        gemmL(stream, xAh, qkvT[l], qkvb[l], qkv, qkvh,
              512, 768, DMODEL, DMODEL, DMODEL, 768, 768,
              1, 1, 0, 0, 0, 0, 0, 0, 1.f, 0);
        // scores[z=(d,h)] = Q @ K^T / sqrt(32), z-offsets via (d,h) composite
        gemmL(stream, qkvh, qkvh + 256, (const float*)0, scores, (_Float16*)0,
              NNODE, NNODE, 32, 768, 768, NNODE, NNODE,
              32, 8, (long)NNODE*768, 32, (long)NNODE*768, 32,
              (long)8*NNODE*NNODE, (long)NNODE*NNODE, isq32, 0);
        softmax128<<<32*NNODE, NNODE, 0, stream>>>(scores, probsh);
        pack_vt<<<32, NNODE, 0, stream>>>(qkv, vth);
        // attn@V -> (4,128,8,32) laid out as (512,256)
        gemmL(stream, probsh, vth, (const float*)0, attno, attnoh,
              NNODE, 32, NNODE, NNODE, NNODE, DMODEL, DMODEL,
              32, 8, (long)8*NNODE*NNODE, (long)NNODE*NNODE,
              (long)8*32*NNODE, (long)32*NNODE,
              (long)NNODE*DMODEL, 32, 1.f, 0);
        // out projection
        gemmL(stream, attnoh, outT[l], outb[l], t0, (_Float16*)0,
              512, DMODEL, DMODEL, DMODEL, DMODEL, DMODEL, DMODEL,
              1, 1, 0, 0, 0, 0, 0, 0, 1.f, 0);
        lnorm_res<<<512, DMODEL, 0, stream>>>(t0, xA, ln1g[l], ln1b[l], xB, xBh);
        // FFN
        gemmL(stream, xBh, ff1T[l], ff1b[l], t1, t1h,
              512, FFD, DMODEL, DMODEL, DMODEL, FFD, FFD,
              1, 1, 0, 0, 0, 0, 0, 0, 1.f, 1);
        gemmL(stream, t1h, ff2T[l], ff2b[l], t0, (_Float16*)0,
              512, DMODEL, FFD, FFD, FFD, DMODEL, DMODEL,
              1, 1, 0, 0, 0, 0, 0, 0, 1.f, 0);
        lnorm_res<<<512, DMODEL, 0, stream>>>(t0, xB, ln2g[l], ln2b[l], xA, xAh);
    }
    bnorm_enc<<<DMODEL, 256, 0, stream>>>(xA, p_bng, p_bnb, venc);

    // ---- edge features + projections ----
    edge_feat<<<NDSET*NPAIR, DMODEL, 0, stream>>>(venc, Eh);
    gemmL(stream, Eh, lineT, line_b, eF, ehPad,
          NDSET*NPAIR, SQD, 512, 512, 512, SQD, 32,
          1, 1, 0, 0, 0, 0, 0, 0, 1.f, 0);
    gemmL(stream, ehPad, linkT, link_b, katt, (_Float16*)0,
          NDSET*NPAIR, 48, 32, 32, 32, 48, 48,
          1, 1, 0, 0, 0, 0, 0, 0, 1.f, 0);
    gemmL(stream, ehPad, linvT, linv_b, vatt, (_Float16*)0,
          NDSET*NPAIR, 32, 32, 32, 32, 32, 32,
          1, 1, 0, 0, 0, 0, 0, 0, 1.f, 0);
    mean_rows<<<NDSET, DMODEL, 0, stream>>>(venc, vmAh);
    gemmL(stream, vmAh, lingT, ling_b, vmC, (_Float16*)0,
          16, 16, DMODEL, DMODEL, DMODEL, 16, 16,
          1, 1, 0, 0, 0, 0, 0, 0, 1.f, 0);

    // ---- autoregressive decode: 127 steps ----
    dec_init<<<BSZ, NNODE, 0, stream>>>(init_nodes, eF, masks, degrees, sbuf,
                                        alive, lens, lsum);
    for (int t = 0; t < TSTEP; ++t) {
        dec_h_k<<<BSZ, SQD, 0, stream>>>(vmC, sbuf, hbuf);
        // self-attn K/V written directly into cache at position t
        gemm16<<<BSZ, SQD, 0, stream>>>(hbuf, SQD, wk_w, wk_b, Ksa + t*SQD, TSTEP*SQD, 0);
        gemm16<<<BSZ, SQD, 0, stream>>>(hbuf, SQD, wv_w, wv_b, Vsa + t*SQD, TSTEP*SQD, 0);
        gemm16<<<BSZ, SQD, 0, stream>>>(hbuf, SQD, wqsa_w, wqsa_b, qb, SQD, 0);
        mha_self<<<BSZ, NHEAD, 0, stream>>>(qb, Ksa, Vsa, t, ob);
        gemm16<<<BSZ, SQD, 0, stream>>>(ob, SQD, w0sa_w, w0sa_b, tb, SQD, 0);
        bnorm_res16<<<SQD, BSZ, 0, stream>>>(hbuf, tb, bn_sa_g, bn_sa_b);
        // cross attention over 16384 candidate edges
        gemm16<<<BSZ, SQD, 0, stream>>>(hbuf, SQD, wqat_w, wqat_b, qb, SQD, 0);
        mha_cross<<<dim3(NHEAD, BSZ), 256, 0, stream>>>(qb, katt, vatt, masks, ob);
        gemm16<<<BSZ, SQD, 0, stream>>>(ob, SQD, w0at_w, w0at_b, tb, SQD, 0);
        bnorm_res16<<<SQD, BSZ, 0, stream>>>(hbuf, tb, bn_at_g, bn_at_b);
        // MLP
        gemm16<<<BSZ, SQD, 0, stream>>>(hbuf, SQD, w1_w, w1_b, mb, SQD, 1);
        gemm16<<<BSZ, SQD, 0, stream>>>(mb, SQD, w2_w, w2_b, tb, SQD, 0);
        bnorm_res16<<<SQD, BSZ, 0, stream>>>(hbuf, tb, bn_mlp_g, bn_mlp_b);
        // pointer head: greedy argmax + logprob
        gemm16<<<BSZ, SQD, 0, stream>>>(hbuf, SQD, wqf_w, wqf_b, qb, SQD, 0);
        final_step<<<BSZ, 256, 0, stream>>>(qb, katt, masks, idxb, logpb);
        dec_update<<<BSZ, NNODE, 0, stream>>>(idxb, logpb, eF, masks, degrees, sbuf,
                                              alive, lens, lsum, out, t, terminal);
    }
    finalize_k<<<1, BSZ, 0, stream>>>(lsum, lens, out);
}
